// TimeCrystalCore_75814762709386
// MI455X (gfx1250) — matrix operand order repaired and checked by reading, not compiled, not run on hardware
//
#include <hip/hip_runtime.h>
#include <math.h>

typedef __attribute__((ext_vector_type(16))) _Float16 v16h;
typedef __attribute__((ext_vector_type(8)))  float    v8f;

#define D_DIM 4096
#define B_DIM 8192
#define TWO_PI_F 6.283185307179586f
#define PHI_F 1.618033f
#define LDK 40   // 32 K halves + 8 pad halves per LDS row

// ---------------- workspace layout (bytes) ----------------
// [0      , 16K )  mean_x        : 4096 f32
// [16K    , 32K )  ncs           : 4096 f32  (new_crystal_state)
// [32K    , 48K )  scalars       : f32 [0]=osc [1]=alpha [2]=beta
// [48K    , 560K)  mean partials : 32 * 4096 f32
// [1M     , 1M+64M) ei_f16       : 8192*4096 _Float16 (enhanced_input)

// -------------------- column mean of x --------------------
__global__ void colmean_partial_k(const float* __restrict__ x,
                                  float* __restrict__ partials) {
    int c = blockIdx.x * 256 + threadIdx.x;       // column
    int r0 = blockIdx.y * (B_DIM / 32);           // row chunk
    float s = 0.f;
    for (int r = 0; r < B_DIM / 32; ++r)
        s += x[(size_t)(r0 + r) * D_DIM + c];
    partials[(size_t)blockIdx.y * D_DIM + c] = s;
}

__global__ void colmean_reduce_k(const float* __restrict__ partials,
                                 float* __restrict__ mean_x) {
    int c = blockIdx.x * 256 + threadIdx.x;
    float s = 0.f;
    for (int p = 0; p < 32; ++p)
        s += partials[(size_t)p * D_DIM + c];
    mean_x[c] = s * (1.0f / (float)B_DIM);
}

// ---------------- scalar chain (1 block) ------------------
__global__ void scalar_k(const float* __restrict__ mean_x,
                         const float* __restrict__ Wf,
                         const float* __restrict__ bf,
                         const float* __restrict__ phase_memory,
                         const float* __restrict__ coh_mom,
                         const int*   __restrict__ time_step,
                         float* __restrict__ scal) {
    __shared__ float red[256];
    int tid = threadIdx.x;
    float p = 0.f;
    for (int i = tid; i < D_DIM; i += 256)
        p += Wf[i] * mean_x[i];
    red[tid] = p;
    __syncthreads();
    for (int s = 128; s > 0; s >>= 1) {
        if (tid < s) red[tid] += red[tid + s];
        __syncthreads();
    }
    if (tid == 0) {
        float ts = (float)(time_step[0] + 1);
        float dotv = red[0] + bf[0];
        float freq_adj = 1.f / (1.f + expf(-dotv));
        float actual_frequency = PHI_F * (1.f + 0.2f * freq_adj);
        float t = ts / 100.f;
        float pp = fmodf(TWO_PI_F * t * actual_frequency, TWO_PI_F);
        float sp = fmodf(pp / PHI_F, TWO_PI_F);
        float osc = 0.3f * (sinf(pp) + 0.3f * cosf(sp));

        // pm = roll(phase_memory,1); pm[0]=pp.  periods[j]=pm[2j]-pm[2j+1]
        float mean = 0.f;
        for (int j = 0; j < 25; ++j) {
            float a = (j == 0) ? pp : phase_memory[2 * j - 1];
            float b = phase_memory[2 * j];
            mean += (a - b);
        }
        mean *= (1.f / 25.f);
        float var = 0.f;
        for (int j = 0; j < 25; ++j) {
            float a = (j == 0) ? pp : phase_memory[2 * j - 1];
            float b = phase_memory[2 * j];
            float d = (a - b) - mean;
            var += d * d;
        }
        float sd = sqrtf(var / 24.f);                      // ddof=1
        float pc = 1.f - sd / TWO_PI_F;
        pc = fminf(fmaxf(pc, 0.f), 1.f);
        float cc = 0.9f * (1.f + 0.2f * pc);
        float stab = fminf(ts / 1000.f, 1.f);
        float coh0 = cc * (1.f + 0.3f * stab);
        float nm = 0.98f * coh_mom[0] + 0.02f * coh0;
        float coh = fminf(fmaxf(nm, 0.9f * 0.8f), 0.9f * 1.5f);
        float tfac = 1.f + fminf(ts / 5000.f, 0.5f);
        scal[0] = osc;
        scal[1] = coh * 1.2f * tfac * 0.3f;                // alpha
        scal[2] = (1.f + coh * 0.2f * tfac) * 0.15f;       // beta
    }
}

// ------- new_crystal_state: Wamp @ mean_x + affine -------
__global__ void matvec_ncs_k(const float* __restrict__ Wamp,
                             const float* __restrict__ mean_x,
                             const float* __restrict__ bamp,
                             const float* __restrict__ crystal_state,
                             const float* __restrict__ crystal_weights,
                             const float* __restrict__ scal,
                             float* __restrict__ ncs) {
    __shared__ float red[128];
    int row = blockIdx.x;
    int tid = threadIdx.x;
    const float* wrow = Wamp + (size_t)row * D_DIM;
    float p = 0.f;
    for (int k = tid; k < D_DIM; k += 128)
        p += wrow[k] * mean_x[k];
    red[tid] = p;
    __syncthreads();
    for (int s = 64; s > 0; s >>= 1) {
        if (tid < s) red[tid] += red[tid + s];
        __syncthreads();
    }
    if (tid == 0) {
        float h = tanhf(red[0] + bamp[row]);
        ncs[row] = 0.98f * crystal_state[row] +
                   0.02f * scal[0] * crystal_weights[row] * (1.f + h);
    }
}

// -------------------- fused WMMA GEMM ---------------------
// C[m,n] = sum_k A[m,k] * W[n,k]   (both row-major, "NT")
// Block tile 128x128, 8 waves (2M x 4N), wave tile 64x32 = 4x2 frags.
template <bool PHASE2>
__global__ __launch_bounds__(256)
void gemm_fused_k(const void* __restrict__ Aptr,     // f32 x (P1) / f16 ei (P2)
                  const float* __restrict__ W,       // Wgate / Wtemp
                  const float* __restrict__ bias,    // bgate / btemp
                  const float* __restrict__ xin,     // x (P1 only)
                  const float* __restrict__ ncs,     // P1 only
                  const float* __restrict__ scal,
                  float* __restrict__ out,
                  _Float16* __restrict__ eih) {
    __shared__ _Float16 Al[2][128 * LDK];
    __shared__ _Float16 Bl[2][128 * LDK];

    const int tid  = threadIdx.x;
    const int lane = tid & 31;
    const int wave = tid >> 5;
    const int waveM = (wave & 1) * 64;
    const int waveN = (wave >> 1) * 32;
    const int l15 = lane & 15;
    const int lh  = lane >> 4;

    const int blockN = blockIdx.x * 128;
    const int blockM = blockIdx.y * 128;

    // staging: thread -> (row, 16-wide K chunk)
    const int sr = tid >> 1;
    const int sk = (tid & 1) * 16;

    v8f acc[4][2];
#pragma unroll
    for (int i = 0; i < 4; ++i)
#pragma unroll
        for (int j = 0; j < 2; ++j)
#pragma unroll
            for (int r = 0; r < 8; ++r) acc[i][j][r] = 0.f;

    float4 ar[4];
    uint4  ah[2];
    float4 br[4];

    // prologue global load (k = 0)
    {
        if constexpr (PHASE2) {
            const uint4* g = (const uint4*)((const _Float16*)Aptr +
                             (size_t)(blockM + sr) * D_DIM + sk);
            ah[0] = g[0]; ah[1] = g[1];
        } else {
            const float4* g = (const float4*)((const float*)Aptr +
                              (size_t)(blockM + sr) * D_DIM + sk);
            ar[0] = g[0]; ar[1] = g[1]; ar[2] = g[2]; ar[3] = g[3];
        }
        const float4* g = (const float4*)(W + (size_t)(blockN + sr) * D_DIM + sk);
        br[0] = g[0]; br[1] = g[1]; br[2] = g[2]; br[3] = g[3];
    }

    int buf = 0;
    for (int k0 = 0; k0 < D_DIM; k0 += 32) {
        // ---- stage regs -> LDS (f16) ----
        {
            _Float16* da = &Al[buf][sr * LDK + sk];
            if constexpr (PHASE2) {
                ((uint4*)da)[0] = ah[0];
                ((uint4*)da)[1] = ah[1];
            } else {
                _Float16 h[16];
#pragma unroll
                for (int i = 0; i < 4; ++i) {
                    h[4 * i + 0] = (_Float16)ar[i].x;
                    h[4 * i + 1] = (_Float16)ar[i].y;
                    h[4 * i + 2] = (_Float16)ar[i].z;
                    h[4 * i + 3] = (_Float16)ar[i].w;
                }
                ((uint4*)da)[0] = *(const uint4*)&h[0];
                ((uint4*)da)[1] = *(const uint4*)&h[8];
            }
            _Float16 hb[16];
#pragma unroll
            for (int i = 0; i < 4; ++i) {
                hb[4 * i + 0] = (_Float16)br[i].x;
                hb[4 * i + 1] = (_Float16)br[i].y;
                hb[4 * i + 2] = (_Float16)br[i].z;
                hb[4 * i + 3] = (_Float16)br[i].w;
            }
            _Float16* db = &Bl[buf][sr * LDK + sk];
            ((uint4*)db)[0] = *(const uint4*)&hb[0];
            ((uint4*)db)[1] = *(const uint4*)&hb[8];
        }
        __syncthreads();

        // ---- kick off next tile's global loads ----
        const int kn = k0 + 32;
        if (kn < D_DIM) {
            if constexpr (PHASE2) {
                const uint4* g = (const uint4*)((const _Float16*)Aptr +
                                 (size_t)(blockM + sr) * D_DIM + kn + sk);
                ah[0] = g[0]; ah[1] = g[1];
            } else {
                const float4* g = (const float4*)((const float*)Aptr +
                                  (size_t)(blockM + sr) * D_DIM + kn + sk);
                ar[0] = g[0]; ar[1] = g[1]; ar[2] = g[2]; ar[3] = g[3];
            }
            const float4* g = (const float4*)(W + (size_t)(blockN + sr) * D_DIM + kn + sk);
            br[0] = g[0]; br[1] = g[1]; br[2] = g[2]; br[3] = g[3];
            if (kn + 32 < D_DIM) {   // L2 prefetch two tiles ahead
                __builtin_prefetch(W + (size_t)(blockN + sr) * D_DIM + kn + 32 + sk, 0, 1);
                if constexpr (!PHASE2)
                    __builtin_prefetch((const float*)Aptr +
                                       (size_t)(blockM + sr) * D_DIM + kn + 32 + sk, 0, 1);
            }
        }

        // ---- B fragments: lane l<16 -> n=l, K 0..15; lane>=16 -> K 16..31 ----
        v16h bfr[2];
#pragma unroll
        for (int fn = 0; fn < 2; ++fn) {
            const _Float16* p = &Bl[buf][(waveN + fn * 16 + l15) * LDK + lh * 16];
            ((uint4*)&bfr[fn])[0] = *(const uint4*)p;
            ((uint4*)&bfr[fn])[1] = *(const uint4*)(p + 8);
        }
        // ---- A fragments: lane l<16 -> m=l, K 0..15; lane>=16 -> K 16..31 ----
#pragma unroll
        for (int fm = 0; fm < 4; ++fm) {
            const _Float16* p = &Al[buf][(waveM + fm * 16 + l15) * LDK + lh * 16];
            v16h a;
            ((uint4*)&a)[0] = *(const uint4*)p;
            ((uint4*)&a)[1] = *(const uint4*)(p + 8);
#pragma unroll
            for (int fn = 0; fn < 2; ++fn)
                acc[fm][fn] = __builtin_amdgcn_wmma_f32_16x16x32_f16(
                    false, a, false, bfr[fn], (short)0, acc[fm][fn], false, false);
        }
        buf ^= 1;
    }

    // -------------------- epilogue --------------------
    const float alpha = scal[1];
    const float beta  = scal[2];
#pragma unroll
    for (int fm = 0; fm < 4; ++fm) {
#pragma unroll
        for (int fn = 0; fn < 2; ++fn) {
#pragma unroll
            for (int r = 0; r < 8; ++r) {
                // C/D layout: VGPR r, lanes 0-15: M=r, N=lane; lanes 16-31: M=r+8
                int row = blockM + waveM + fm * 16 + r + lh * 8;
                int col = blockN + waveN + fn * 16 + l15;
                size_t idx = (size_t)row * D_DIM + col;
                float v = acc[fm][fn][r] + bias[col];
                if constexpr (!PHASE2) {
                    float g = 1.f / (1.f + expf(-v));           // sigmoid gate
                    float ei = xin[idx] + g * ncs[col] * alpha; // enhanced_input
                    out[idx] = ei;
                    eih[idx] = (_Float16)ei;
                } else {
                    float tf = tanhf(v);                        // temporal feature
                    out[idx] = out[idx] + tf * beta;            // final output
                }
            }
        }
    }
}

// ------------------------ launcher ------------------------
extern "C" void kernel_launch(void* const* d_in, const int* in_sizes, int n_in,
                              void* d_out, int out_size, void* d_ws, size_t ws_size,
                              hipStream_t stream) {
    const float* x               = (const float*)d_in[0];
    const float* crystal_state   = (const float*)d_in[1];
    const float* phase_memory    = (const float*)d_in[2];
    const float* coh_mom         = (const float*)d_in[3];
    const float* crystal_weights = (const float*)d_in[4];
    const float* Wf              = (const float*)d_in[5];
    const float* bf              = (const float*)d_in[6];
    const float* Wamp            = (const float*)d_in[7];
    const float* bamp            = (const float*)d_in[8];
    const float* Wtemp           = (const float*)d_in[9];
    const float* btemp           = (const float*)d_in[10];
    const float* Wgate           = (const float*)d_in[11];
    const float* bgate           = (const float*)d_in[12];
    const int*   time_step       = (const int*)d_in[13];

    char* ws = (char*)d_ws;
    float*     mean_x   = (float*)(ws);
    float*     ncs      = (float*)(ws + (16 << 10));
    float*     scal     = (float*)(ws + (32 << 10));
    float*     partials = (float*)(ws + (48 << 10));
    _Float16*  eih      = (_Float16*)(ws + (1 << 20));

    float* out = (float*)d_out;

    colmean_partial_k<<<dim3(D_DIM / 256, 32), 256, 0, stream>>>(x, partials);
    colmean_reduce_k<<<D_DIM / 256, 256, 0, stream>>>(partials, mean_x);
    scalar_k<<<1, 256, 0, stream>>>(mean_x, Wf, bf, phase_memory, coh_mom,
                                    time_step, scal);
    matvec_ncs_k<<<D_DIM, 128, 0, stream>>>(Wamp, mean_x, bamp, crystal_state,
                                            crystal_weights, scal, ncs);

    dim3 grid(D_DIM / 128, B_DIM / 128);  // (N tiles, M tiles)
    gemm_fused_k<false><<<grid, 256, 0, stream>>>(x, Wgate, bgate, x, ncs,
                                                  scal, out, eih);
    gemm_fused_k<true><<<grid, 256, 0, stream>>>(eih, Wtemp, btemp, nullptr,
                                                 nullptr, scal, out, eih);
}
